// GCN_82334523064419
// MI455X (gfx1250) — matrix operand order repaired and checked by reading, not compiled, not run on hardware
//
#include <hip/hip_runtime.h>
#include <hip/hip_bf16.h>
#include <stdint.h>

// ---------------------------------------------------------------------------
// GCN: out = relu(A_norm @ (x@W1) + b1) -> relu(A_norm @ (h@W2) + b2) @ Wfc + bfc
// A_norm = D^-1/2 (A + I) D^-1/2.
// Dense transforms: v_wmma_f32_16x16x32_bf16, one wave = 16x64 row-block
// (4 independent accumulator chains, A-fragment reused across 4 WMMAs).
// Weights pre-packed once into bf16 fragment layout -> contiguous b128 loads.
// Sparse aggregation: f32 atomics (h = 25.6MB, resident in 192MB L2).
// ---------------------------------------------------------------------------

typedef __attribute__((ext_vector_type(16))) __bf16 v16bf;
typedef __attribute__((ext_vector_type(8)))  float  v8f;

#define FIN  256
#define HID  64
#define NCLS 11
#define NT   4      // N-tiles per wave (HID/16)

// ---------------- degree / normalization ----------------
__global__ void deg_init_kernel(float* __restrict__ deg, int n) {
    int i = blockIdx.x * blockDim.x + threadIdx.x;
    if (i < n) deg[i] = 1.0f;  // self loop
}

__global__ void deg_count_kernel(const int* __restrict__ dst, float* __restrict__ deg, int E) {
    int e = blockIdx.x * blockDim.x + threadIdx.x;
    if (e < E) atomicAdd(&deg[dst[e]], 1.0f);
}

__global__ void deg_rsqrt_kernel(float* __restrict__ deg, int n) {
    int i = blockIdx.x * blockDim.x + threadIdx.x;
    if (i < n) deg[i] = rsqrtf(deg[i]);
}

// ---------------- weight repack: W[K,64] f32 -> bf16 B-fragments ----------------
// Fragment layout (32x16 bf16 B, contiguous k order used here, same as A): lanes 0-15 hold K=k0..k0+15 for
// col N=lane; lanes 16-31 hold K=k0+16..k0+31 for col N=lane-16.
// pack index: ((kstep*NT + t)*32 + lane)*16 + e  -> 32B contiguous per lane.
__global__ void pack_b_kernel(const float* __restrict__ W, __bf16* __restrict__ pack, int K) {
    int i = blockIdx.x * blockDim.x + threadIdx.x;     // one thread = one lane-frag
    int total = (K >> 5) * NT * 32;
    if (i >= total) return;
    int lane  = i & 31;
    int t     = (i >> 5) & (NT - 1);
    int kstep = i >> 7;                                 // /(32*NT), NT==4
    int hl = lane >> 4, l = lane & 15;
    int n = t * 16 + l;
    int kbase = (kstep << 5) + hl * 16;
    __bf16* dst = pack + (size_t)i * 16;
#pragma unroll
    for (int e = 0; e < 16; ++e) dst[e] = (__bf16)W[(size_t)(kbase + e) * HID + n];
}

// ---------------- dense transform: C[M,64] = A[M,K] @ W[K,64] ----------------
// One wave per 16-row block: A-frag converted once per K-step, 4 WMMAs issue on
// 4 independent accumulators (no wmma->wmma RAW chain between them).
__global__ void gemm_bf16_wmma_kernel(const float* __restrict__ A,
                                      const __bf16* __restrict__ Bpack,
                                      float* __restrict__ C, int M, int K) {
    const int lane = threadIdx.x & 31;
    const int wave = blockIdx.x * (blockDim.x >> 5) + (threadIdx.x >> 5);
    if (wave * 16 >= M) return;            // wave-uniform; EXEC stays all-1s for WMMA

    const int hl = lane >> 4;              // half-wave 0/1
    const int l  = lane & 15;
    const long long arow = (long long)(wave * 16 + l);

    v8f acc[NT] = {};
    const int ksteps = K >> 5;
    for (int ks = 0; ks < ksteps; ++ks) {
        // A-frag (16-bit A 16x32): elems 0..15 = K k0+hl*16+0..15
        const float* ap = A + arow * K + (ks << 5) + hl * 16;
        v16bf af;
#pragma unroll
        for (int e = 0; e < 8; ++e) af[e]     = (__bf16)ap[e];
#pragma unroll
        for (int e = 0; e < 8; ++e) af[8 + e] = (__bf16)ap[8 + e];

        const v16bf* bp = (const v16bf*)(Bpack + ((size_t)(ks * NT) * 32 + lane) * 16);
#pragma unroll
        for (int t = 0; t < NT; ++t) {
            v16bf bfrag = bp[t * 32];      // contiguous 32B per lane, coalesced
            acc[t] = __builtin_amdgcn_wmma_f32_16x16x32_bf16(
                false, af, false, bfrag, (short)0, acc[t], false, false);
        }
    }

    // C/D layout: vgpr j, lanes 0-15 -> row j, lanes 16-31 -> row j+8
#pragma unroll
    for (int t = 0; t < NT; ++t) {
        float* cp = C + (long long)(wave * 16 + hl * 8) * HID + t * 16 + l;
#pragma unroll
        for (int j = 0; j < 8; ++j) cp[(long long)j * HID] = acc[t][j];
    }
}

// ---------------- sparse aggregation ----------------
// out[i,:] = dinv[i]^2 * h[i,:]   (self-loop term, initializes accumulator)
__global__ void agg_selfloop_init_kernel(const float* __restrict__ h,
                                         const float* __restrict__ dinv,
                                         float* __restrict__ out, int n) {
    int i = blockIdx.x * blockDim.x + threadIdx.x;  // over n*HID
    if (i < n * HID) {
        float d = dinv[i >> 6];  // HID == 64
        out[i] = d * d * h[i];
    }
}

// out[dst,:] += dinv[src]*dinv[dst] * h[src,:]   (8 threads/edge, 8 floats each)
__global__ void agg_edges_kernel(const int* __restrict__ src,
                                 const int* __restrict__ dst,
                                 const float* __restrict__ dinv,
                                 const float* __restrict__ h,
                                 float* __restrict__ out, int E) {
    long long t = (long long)blockIdx.x * blockDim.x + threadIdx.x;
    if (t >= (long long)E * 8) return;
    int e = (int)(t >> 3);
    int f = (int)(t & 7) << 3;
    int s = src[e], d = dst[e];
    float nrm = dinv[s] * dinv[d];
    const float4* hp = (const float4*)(h + (long long)s * HID + f);
    float4 h0 = hp[0], h1 = hp[1];
    float* op = out + (long long)d * HID + f;
    atomicAdd(op + 0, nrm * h0.x);
    atomicAdd(op + 1, nrm * h0.y);
    atomicAdd(op + 2, nrm * h0.z);
    atomicAdd(op + 3, nrm * h0.w);
    atomicAdd(op + 4, nrm * h1.x);
    atomicAdd(op + 5, nrm * h1.y);
    atomicAdd(op + 6, nrm * h1.z);
    atomicAdd(op + 7, nrm * h1.w);
}

__global__ void bias_relu_kernel(float* __restrict__ out, const float* __restrict__ b, int n) {
    int i = blockIdx.x * blockDim.x + threadIdx.x;  // over n*HID
    if (i < n * HID) {
        float v = out[i] + b[i & (HID - 1)];
        out[i] = v > 0.0f ? v : 0.0f;
    }
}

// ---------------- final classifier: out[N,11] = h[N,64] @ Wfc[64,11] + bfc ----
__global__ void fc_kernel(const float* __restrict__ h, const float* __restrict__ W,
                          const float* __restrict__ b, float* __restrict__ out, int n) {
    int i = blockIdx.x * blockDim.x + threadIdx.x;  // over n*NCLS
    if (i >= n * NCLS) return;
    int node = i / NCLS, c = i % NCLS;
    const float* hp = h + (long long)node * HID;
    float s = b[c];
#pragma unroll
    for (int k = 0; k < HID; ++k) s += hp[k] * W[k * NCLS + c];
    out[i] = s;
}

// ---------------------------------------------------------------------------
extern "C" void kernel_launch(void* const* d_in, const int* in_sizes, int n_in,
                              void* d_out, int out_size, void* d_ws, size_t ws_size,
                              hipStream_t stream) {
    const float* x    = (const float*)d_in[0];       // [N, 256]
    const int*   eidx = (const int*)d_in[1];         // [2, E]: row0 src, row1 dst
    const float* W1   = (const float*)d_in[2];       // [256, 64]
    const float* b1   = (const float*)d_in[3];       // [64]
    const float* W2   = (const float*)d_in[4];       // [64, 64]
    const float* b2   = (const float*)d_in[5];       // [64]
    const float* Wfc  = (const float*)d_in[6];       // [64, 11]
    const float* bfc  = (const float*)d_in[7];       // [11]
    float* out = (float*)d_out;                      // [N, 11]

    const int N = in_sizes[0] / FIN;
    const int E = in_sizes[1] / 2;
    const int* src = eidx;
    const int* dst = eidx + E;

    // workspace: deg | htmp (N*64 f32) | hagg (N*64 f32) | bpack1 | bpack2
    size_t off = ((size_t)N * 4 + 255) & ~(size_t)255;
    float* deg  = (float*)d_ws;
    float* htmp = (float*)((char*)d_ws + off);
    off += (size_t)N * HID * 4;
    float* hagg = (float*)((char*)d_ws + off);
    off += (size_t)N * HID * 4;
    __bf16* bpack1 = (__bf16*)((char*)d_ws + off);   // (256/32)*4*32*16 bf16 = 32KB
    off += (size_t)(FIN >> 5) * NT * 32 * 16 * 2;
    off = (off + 255) & ~(size_t)255;
    __bf16* bpack2 = (__bf16*)((char*)d_ws + off);   // (64/32)*4*32*16 bf16 = 8KB

    const int B256 = 256;
    const int gN    = (N + B256 - 1) / B256;
    const int gE    = (E + B256 - 1) / B256;
    const int gNH   = (N * HID + B256 - 1) / B256;
    const int gEdge = (int)(((long long)E * 8 + B256 - 1) / B256);
    const int gFC   = (N * NCLS + B256 - 1) / B256;
    const int gPk1  = ((FIN >> 5) * NT * 32 + B256 - 1) / B256;
    const int gPk2  = ((HID >> 5) * NT * 32 + B256 - 1) / B256;
    const int gGemm = ((N / 16) + 7) / 8;            // 8 waves per 256-thread block

    // normalization: deg = 1 + indeg;  dinv = rsqrt(deg); weight repacks
    deg_init_kernel<<<gN, B256, 0, stream>>>(deg, N);
    deg_count_kernel<<<gE, B256, 0, stream>>>(dst, deg, E);
    deg_rsqrt_kernel<<<gN, B256, 0, stream>>>(deg, N);
    pack_b_kernel<<<gPk1, B256, 0, stream>>>(W1, bpack1, FIN);
    pack_b_kernel<<<gPk2, B256, 0, stream>>>(W2, bpack2, HID);

    // ---- layer 1 ----
    gemm_bf16_wmma_kernel<<<gGemm, B256, 0, stream>>>(x, bpack1, htmp, N, FIN);
    agg_selfloop_init_kernel<<<gNH, B256, 0, stream>>>(htmp, deg, hagg, N);
    agg_edges_kernel<<<gEdge, B256, 0, stream>>>(src, dst, deg, htmp, hagg, E);
    bias_relu_kernel<<<gNH, B256, 0, stream>>>(hagg, b1, N);

    // ---- layer 2 (reuse htmp/hagg ping-pong) ----
    gemm_bf16_wmma_kernel<<<gGemm, B256, 0, stream>>>(hagg, bpack2, htmp, N, HID);
    agg_selfloop_init_kernel<<<gNH, B256, 0, stream>>>(htmp, deg, hagg, N);
    agg_edges_kernel<<<gEdge, B256, 0, stream>>>(src, dst, deg, htmp, hagg, E);
    bias_relu_kernel<<<gNH, B256, 0, stream>>>(hagg, b2, N);

    // ---- classifier ----
    fc_kernel<<<gFC, B256, 0, stream>>>(hagg, Wfc, bfc, out, N);
}
